// EMAQuantizer_58712202936470
// MI455X (gfx1250) — compile-verified, matrix operand order checked by reading, not run on hardware
//
#include <hip/hip_runtime.h>

// ---------------- problem constants ----------------
#define NUM_EMB   1024
#define EMB_DIM   64
#define BATCH     32
#define T_LEN     2048
#define NPTS      (BATCH * T_LEN)          // 65536
#define ROWS_PER_BLOCK 128
#define NBLOCKS   (NPTS / ROWS_PER_BLOCK)  // 512

// ---------------- output layout (flat f32, return order) ----------------
#define Q_SIZE    (BATCH * EMB_DIM * T_LEN)   // 4194304
#define CL_OFF    (Q_SIZE)                    // commitment loss scalar
#define PP_OFF    (Q_SIZE + 1)                // perplexity scalar
#define AVG_OFF   (Q_SIZE + 2)                // avg_probs[1024]
#define IDX_OFF   (Q_SIZE + 2 + NUM_EMB)      // encoding_idx[65536] as float
#define UL_OFF    (IDX_OFF + NPTS)            // usage loss scalar

// ---------------- workspace layout (bytes) ----------------
#define EHI_OFF   0                           // bf16 (ushort) [1024*64]
#define ELO_OFF   131072                      // bf16 (ushort) [1024*64]
#define ESQ_OFF   262144                      // f32  [1024]
#define CNT_OFF   266240                      // u32  [1024]
#define PART_OFF  270336                      // f32  [512] per-block loss partials

typedef __attribute__((ext_vector_type(16))) __bf16 v16bf;
typedef __attribute__((ext_vector_type(8)))  __bf16 v8bf;
typedef __attribute__((ext_vector_type(8)))  float  v8f;

__device__ __forceinline__ unsigned short f2bfu(float f) {
    unsigned u = __float_as_uint(f);
    u = u + 0x7FFFu + ((u >> 16) & 1u);   // round-to-nearest-even
    return (unsigned short)(u >> 16);
}
__device__ __forceinline__ float bfu2f(unsigned short h) {
    return __uint_as_float(((unsigned)h) << 16);
}
__device__ __forceinline__ __bf16 u2bf(unsigned short h) {
    __bf16 b; __builtin_memcpy(&b, &h, 2); return b;
}

// ============ kernel 0: split codebook into bf16 hi/lo, ||e||^2, zero counts ============
__global__ __launch_bounds__(256)
void vq_prep(const float* __restrict__ embed,
             unsigned short* __restrict__ ehi, unsigned short* __restrict__ elo,
             float* __restrict__ esq, unsigned* __restrict__ cnt) {
    int k = blockIdx.x * 256 + threadIdx.x;
    if (k >= NUM_EMB) return;
    float ss = 0.f;
    #pragma unroll 8
    for (int j = 0; j < EMB_DIM; ++j) {
        float v = embed[k * EMB_DIM + j];
        ss += v * v;
        unsigned short h = f2bfu(v);
        ehi[k * EMB_DIM + j] = h;
        elo[k * EMB_DIM + j] = f2bfu(v - bfu2f(h));
    }
    esq[k] = ss;
    cnt[k] = 0u;
}

#define WMMA_BF16(A, B, C) \
    __builtin_amdgcn_wmma_f32_16x16x32_bf16(false, (A), false, (B), (short)0, (C), false, false)

#define SHUF16(a, b) __builtin_shufflevector((a), (b), 0,1,2,3,4,5,6,7,8,9,10,11,12,13,14,15)

// ============ kernel 1: nearest-code search (WMMA) + quantize + partial losses ============
__global__ __launch_bounds__(256)
void vq_main(const float* __restrict__ x, const float* __restrict__ embed,
             const unsigned short* __restrict__ ehi, const unsigned short* __restrict__ elo,
             const float* __restrict__ esq, unsigned* __restrict__ cnt,
             float* __restrict__ out, float* __restrict__ part) {
    __shared__ float lds_x[EMB_DIM * ROWS_PER_BLOCK];   // [d][tt] 32 KB
    __shared__ float lds_esq[NUM_EMB];                  // 4 KB
    __shared__ int   lds_bidx[ROWS_PER_BLOCK];
    __shared__ float lds_red[256];

    const int tid     = threadIdx.x;
    const int rowbase = blockIdx.x * ROWS_PER_BLOCK;    // always within one batch b
    const int b       = rowbase >> 11;
    const int t0      = rowbase & (T_LEN - 1);
    const int xbase   = b * EMB_DIM * T_LEN + t0;

    // --- stage x tile (fixes the [b,d,t] stride) + esq into LDS, coalesced ---
    #pragma unroll
    for (int i = 0; i < 32; ++i) {
        int idx = tid + i * 256;
        int j = idx >> 7, tt = idx & 127;
        lds_x[idx] = x[xbase + j * T_LEN + tt];
    }
    #pragma unroll
    for (int i = 0; i < 4; ++i) lds_esq[tid + i * 256] = esq[tid + i * 256];
    __syncthreads();

    const int lane = tid & 31;
    const int wave = tid >> 5;
    const int m    = lane & 15;           // A-row / B-col within 16-group
    const int loff = (lane >> 4) << 3;    // k offset 0 or 8 for upper lane half
    const int tt   = wave * 16 + m;       // this lane's x row within block

    // --- build split-bf16 A fragments (16 rows x 64 d), reused for all 64 code tiles ---
    v16bf Ah[2], Al[2];
    #pragma unroll
    for (int c = 0; c < 2; ++c) {
        #pragma unroll
        for (int e = 0; e < 16; ++e) {
            int k = c * 32 + loff + (e < 8 ? e : e + 8);
            float v = lds_x[k * ROWS_PER_BLOCK + tt];
            unsigned short h = f2bfu(v);
            Ah[c][e] = u2bf(h);
            Al[c][e] = u2bf(f2bfu(v - bfu2f(h)));
        }
    }

    float best[8]; int bidx[8];
    #pragma unroll
    for (int v = 0; v < 8; ++v) { best[v] = 3.4e38f; bidx[v] = 0; }

    // --- sweep all 1024 codes in 16-wide tiles: 6 bf16 WMMAs per tile ---
    for (int tile = 0; tile < NUM_EMB / 16; ++tile) {
        const int cb = tile * 16;
        const v8bf* ph = (const v8bf*)(ehi + (cb + m) * EMB_DIM + loff);
        const v8bf* pl = (const v8bf*)(elo + (cb + m) * EMB_DIM + loff);
        v16bf Bh0 = SHUF16(ph[0], ph[2]);   // k = loff..+7, loff+16..+23
        v16bf Bh1 = SHUF16(ph[4], ph[6]);   // + 32
        v16bf Bl0 = SHUF16(pl[0], pl[2]);
        v16bf Bl1 = SHUF16(pl[4], pl[6]);

        v8f acc = {};
        acc = WMMA_BF16(Ah[0], Bh0, acc);
        acc = WMMA_BF16(Ah[1], Bh1, acc);
        acc = WMMA_BF16(Ah[0], Bl0, acc);
        acc = WMMA_BF16(Ah[1], Bl1, acc);
        acc = WMMA_BF16(Al[0], Bh0, acc);
        acc = WMMA_BF16(Al[1], Bh1, acc);

        // argmin only needs esq - 2*dot (||x||^2 is constant per row)
        const float es = lds_esq[cb + m];
        const int code = cb + m;
        #pragma unroll
        for (int v = 0; v < 8; ++v) {
            float s = es - 2.0f * acc[v];
            if (s < best[v]) { best[v] = s; bidx[v] = code; }
        }
    }

    // --- min+index reduce across the 16 lanes sharing each output row ---
    #pragma unroll
    for (int v = 0; v < 8; ++v) {
        #pragma unroll
        for (int off = 1; off < 16; off <<= 1) {
            float os = __shfl_xor(best[v], off, 32);
            int   oi = __shfl_xor(bidx[v], off, 32);
            if (os < best[v] || (os == best[v] && oi < bidx[v])) { best[v] = os; bidx[v] = oi; }
        }
    }

    if (lane == 0 || lane == 16) {          // lane0: rows 0-7, lane16: rows 8-15 of wave tile
        const int rsel = (lane == 16) ? 8 : 0;
        #pragma unroll
        for (int v = 0; v < 8; ++v) {
            int rl = wave * 16 + rsel + v;
            int code = bidx[v];
            lds_bidx[rl] = code;
            out[IDX_OFF + rowbase + rl] = (float)code;
            atomicAdd(&cnt[code], 1u);
        }
    }
    __syncthreads();

    // --- gather quantized (== quantized_st in fwd) + commitment partial ---
    float lsum = 0.f;
    #pragma unroll
    for (int i = 0; i < 32; ++i) {
        int idx = tid + i * 256;
        int j = idx >> 7, ttt = idx & 127;
        float q  = embed[lds_bidx[ttt] * EMB_DIM + j];
        float xv = lds_x[idx];
        float d  = q - xv;
        lsum += d * d;
        out[xbase + j * T_LEN + ttt] = q;
    }
    lds_red[tid] = lsum;
    __syncthreads();
    #pragma unroll
    for (int s = 128; s > 0; s >>= 1) {
        if (tid < s) lds_red[tid] += lds_red[tid + s];
        __syncthreads();
    }
    if (tid == 0) part[blockIdx.x] = lds_red[0];
}

// ============ kernel 2: finalize scalars + avg_probs (deterministic) ============
__global__ __launch_bounds__(1024)
void vq_final(const unsigned* __restrict__ cnt, const float* __restrict__ part,
              float* __restrict__ out) {
    __shared__ float red[1024];
    const int tid = threadIdx.x;

    // commitment loss
    red[tid] = (tid < NBLOCKS) ? part[tid] : 0.f;
    __syncthreads();
    for (int s = 512; s > 0; s >>= 1) { if (tid < s) red[tid] += red[tid + s]; __syncthreads(); }
    if (tid == 0) out[CL_OFF] = 0.25f * red[0] / (float)Q_SIZE;
    __syncthreads();

    // avg_probs + perplexity
    float p = (float)cnt[tid] / (float)NPTS;
    out[AVG_OFF + tid] = p;
    red[tid] = p * logf(p + 1e-10f);
    __syncthreads();
    for (int s = 512; s > 0; s >>= 1) { if (tid < s) red[tid] += red[tid + s]; __syncthreads(); }
    if (tid == 0) out[PP_OFF] = expf(-red[0]);
    __syncthreads();

    // usage loss
    red[tid] = p * logf(p * (float)NUM_EMB + 1e-10f);
    __syncthreads();
    for (int s = 512; s > 0; s >>= 1) { if (tid < s) red[tid] += red[tid + s]; __syncthreads(); }
    if (tid == 0) out[UL_OFF] = red[0];
}

extern "C" void kernel_launch(void* const* d_in, const int* in_sizes, int n_in,
                              void* d_out, int out_size, void* d_ws, size_t ws_size,
                              hipStream_t stream) {
    const float* x     = (const float*)d_in[0];   // [32, 64, 2048]
    const float* embed = (const float*)d_in[1];   // [1024, 64]
    float* out = (float*)d_out;

    char* ws = (char*)d_ws;
    unsigned short* ehi = (unsigned short*)(ws + EHI_OFF);
    unsigned short* elo = (unsigned short*)(ws + ELO_OFF);
    float*          esq = (float*)(ws + ESQ_OFF);
    unsigned*       cnt = (unsigned*)(ws + CNT_OFF);
    float*          prt = (float*)(ws + PART_OFF);

    vq_prep <<<NUM_EMB / 256, 256, 0, stream>>>(embed, ehi, elo, esq, cnt);
    vq_main <<<NBLOCKS,       256, 0, stream>>>(x, embed, ehi, elo, esq, cnt, out, prt);
    vq_final<<<1,            1024, 0, stream>>>(cnt, prt, out);
}
